// SparseAutoencoder_37177236914948
// MI455X (gfx1250) — compile-verified, matrix operand order checked by reading, not run on hardware
//
#include <hip/hip_runtime.h>
#include <hip/hip_bf16.h>
#include <stdint.h>

// Problem sizes (fixed by the reference)
#define BATCH   4096
#define IN_DIM  4096
#define LAT_DIM 32768
#define TOPK    64

// GEMM tiling
#define BM 128
#define BN 128
#define BK 32
#define LDST 40   // padded LDS k-stride (bf16 elems): 80B rows -> conflict-free halves

typedef __attribute__((ext_vector_type(16))) __bf16    v16bf;
typedef __attribute__((ext_vector_type(8)))  float     v8f;
typedef __attribute__((ext_vector_type(4)))  unsigned  u32x4;   // trivial (POD) 16B vector
typedef __attribute__((ext_vector_type(4)))  float     f32x4;   // POD float4 (for NT access)

union FragBF { u32x4 q[2]; v16bf v; };

// ---------- helpers ----------
__device__ __forceinline__ unsigned short f2bf(float f) {
    unsigned u = __float_as_uint(f);
    u += 0x7FFFu + ((u >> 16) & 1u);      // round-to-nearest-even
    return (unsigned short)(u >> 16);
}
__device__ __forceinline__ unsigned pack2(float a, float b) {
    return (unsigned)f2bf(a) | ((unsigned)f2bf(b) << 16);
}
// monotonic uint key: order(key) == order(float)
__device__ __forceinline__ unsigned f2mono(float f) {
    unsigned u = __float_as_uint(f);
    return (u & 0x80000000u) ? ~u : (u | 0x80000000u);
}
__device__ __forceinline__ float mono2f(unsigned k) {
    unsigned u = (k & 0x80000000u) ? (k & 0x7FFFFFFFu) : ~k;
    return __uint_as_float(u);
}

// ============================================================
// Kernel A: z_pre[b,l] = sum_d h[b,d]*W_enc[l,d] + b_enc[l]
// bf16 WMMA, f32 accumulate. Writes z_pre into the z output region.
// Grid: x = M (fast) so concurrent blocks share the same W_enc strip;
// h (64 MB) stays L2-resident across the whole N sweep.
// ============================================================
__global__ __launch_bounds__(256) void sae_encode_gemm(
    const float* __restrict__ h, const float* __restrict__ W_enc,
    const float* __restrict__ b_enc, float* __restrict__ z)
{
    __shared__ __attribute__((aligned(16))) unsigned short As[BM * LDST];
    __shared__ __attribute__((aligned(16))) unsigned short Bs[BN * LDST];

    const int tid      = threadIdx.x;
    const int lane     = tid & 31;
    const int wave     = tid >> 5;       // 0..7
    const int waveM    = wave & 3;       // 4 waves along M (32 rows each)
    const int waveN    = wave >> 2;      // 2 waves along N (64 cols each)
    const int laneRow  = lane & 15;
    const int laneHalf = lane >> 4;
    const int kbase    = laneHalf * 8;   // ISA 16-bit A/B striping: half-lane K offset

    const int m0 = blockIdx.x * BM;      // batch rows   (fast grid dim)
    const int n0 = blockIdx.y * BN;      // latent cols  (slow grid dim)

    // staging: thread -> (row 0..127, 16-float segment 0..1)
    const int sRow = tid >> 1;
    const int sSeg = (tid & 1) * 16;

    v8f acc[2][4] = {};

    for (int k0 = 0; k0 < IN_DIM; k0 += BK) {
        __syncthreads();
        // ---- stage A tile (h rows), f32 -> bf16 ----
        {
            const float4* g = reinterpret_cast<const float4*>(
                h + (size_t)(m0 + sRow) * IN_DIM + k0 + sSeg);
            float4 f0 = g[0], f1 = g[1], f2 = g[2], f3 = g[3];
            u32x4 p0, p1;
            p0.x = pack2(f0.x, f0.y); p0.y = pack2(f0.z, f0.w);
            p0.z = pack2(f1.x, f1.y); p0.w = pack2(f1.z, f1.w);
            p1.x = pack2(f2.x, f2.y); p1.y = pack2(f2.z, f2.w);
            p1.z = pack2(f3.x, f3.y); p1.w = pack2(f3.z, f3.w);
            u32x4* dst = reinterpret_cast<u32x4*>(&As[sRow * LDST + sSeg]);
            dst[0] = p0; dst[1] = p1;
        }
        // ---- stage B tile (W_enc rows = latent), f32 -> bf16 ----
        {
            const float4* g = reinterpret_cast<const float4*>(
                W_enc + (size_t)(n0 + sRow) * IN_DIM + k0 + sSeg);
            float4 f0 = g[0], f1 = g[1], f2 = g[2], f3 = g[3];
            u32x4 p0, p1;
            p0.x = pack2(f0.x, f0.y); p0.y = pack2(f0.z, f0.w);
            p0.z = pack2(f1.x, f1.y); p0.w = pack2(f1.z, f1.w);
            p1.x = pack2(f2.x, f2.y); p1.y = pack2(f2.z, f2.w);
            p1.z = pack2(f3.x, f3.y); p1.w = pack2(f3.z, f3.w);
            u32x4* dst = reinterpret_cast<u32x4*>(&Bs[sRow * LDST + sSeg]);
            dst[0] = p0; dst[1] = p1;
        }
        __syncthreads();

        // ---- fragments: two contiguous 16B halves per lane (K kbase..+7, 16+kbase..+7) ----
        FragBF a[2], b[4];
        #pragma unroll
        for (int mt = 0; mt < 2; ++mt) {
            const int r = (waveM * 32 + mt * 16 + laneRow) * LDST;
            a[mt].q[0] = *reinterpret_cast<const u32x4*>(&As[r + kbase]);
            a[mt].q[1] = *reinterpret_cast<const u32x4*>(&As[r + 16 + kbase]);
        }
        #pragma unroll
        for (int nt = 0; nt < 4; ++nt) {
            const int r = (waveN * 64 + nt * 16 + laneRow) * LDST;
            b[nt].q[0] = *reinterpret_cast<const u32x4*>(&Bs[r + kbase]);
            b[nt].q[1] = *reinterpret_cast<const u32x4*>(&Bs[r + 16 + kbase]);
        }

        #pragma unroll
        for (int mt = 0; mt < 2; ++mt)
            #pragma unroll
            for (int nt = 0; nt < 4; ++nt)
                acc[mt][nt] = __builtin_amdgcn_wmma_f32_16x16x32_bf16(
                    false, a[mt].v, false, b[nt].v,
                    (short)0, acc[mt][nt], false, false);
    }

    // ---- epilogue: C layout lane=(N), vgpr r -> M = r + 8*laneHalf ----
    // z_pre is a 512MB stream: non-temporal stores keep h/W_enc hot in L2.
    #pragma unroll
    for (int nt = 0; nt < 4; ++nt) {
        const int col = n0 + waveN * 64 + nt * 16 + laneRow;
        const float be = b_enc[col];
        #pragma unroll
        for (int mt = 0; mt < 2; ++mt) {
            const int rowBase = m0 + waveM * 32 + mt * 16 + laneHalf * 8;
            #pragma unroll
            for (int r = 0; r < 8; ++r)
                __builtin_nontemporal_store(acc[mt][nt][r] + be,
                    z + (size_t)(rowBase + r) * LAT_DIM + col);
        }
    }
}

// ============================================================
// Kernel B: per-row exact top-64 (radix select on monotonic keys),
// in-place rewrite of z, emit (idx,val) pairs for the decoder.
// z row is streamed once (read+write) -> non-temporal.
// ============================================================
__global__ __launch_bounds__(1024) void sae_topk(
    float* __restrict__ z, int* __restrict__ nzIdx, float* __restrict__ nzVal)
{
    const int row  = blockIdx.x;
    const int tid  = threadIdx.x;
    const int lane = tid & 31;
    const int wv   = tid >> 5;
    float* zrow = z + (size_t)row * LAT_DIM;

    __shared__ int warpSums[32];
    __shared__ int sTotal;
    __shared__ int tieCnt, slotCnt;

    unsigned keys[32];
    const f32x4* zr4 = reinterpret_cast<const f32x4*>(zrow);
    #pragma unroll
    for (int j = 0; j < 8; ++j) {
        f32x4 f = __builtin_nontemporal_load(zr4 + tid + 1024 * j);
        keys[j * 4 + 0] = f2mono(f.x); keys[j * 4 + 1] = f2mono(f.y);
        keys[j * 4 + 2] = f2mono(f.z); keys[j * 4 + 3] = f2mono(f.w);
    }

    // MSB-first radix select: prefix -> exact 64th-largest key
    unsigned prefix = 0;
    int kneed = TOPK;
    for (int bit = 31; bit >= 0; --bit) {
        const unsigned mask = ~((1u << bit) - 1u);
        const unsigned test = prefix | (1u << bit);
        int c = 0;
        #pragma unroll
        for (int e = 0; e < 32; ++e) c += ((keys[e] & mask) == test);
        for (int off = 16; off > 0; off >>= 1) c += __shfl_down(c, off, 32);
        if (lane == 0) warpSums[wv] = c;
        __syncthreads();
        if (tid < 32) {
            int s = warpSums[tid];
            for (int off = 16; off > 0; off >>= 1) s += __shfl_down(s, off, 32);
            if (tid == 0) sTotal = s;
        }
        __syncthreads();
        const int total = sTotal;
        if (total >= kneed) prefix = test; else kneed -= total;
    }

    if (tid == 0) { tieCnt = 0; slotCnt = 0; }
    __syncthreads();
    const unsigned kthKey = prefix;   // exact K-th largest key; kneed = tie budget

    #pragma unroll
    for (int j = 0; j < 8; ++j) {
        float ov[4];
        #pragma unroll
        for (int c4 = 0; c4 < 4; ++c4) {
            const unsigned key = keys[j * 4 + c4];
            const int e = (tid + 1024 * j) * 4 + c4;
            bool keep = false;
            if (key > kthKey)       keep = true;
            else if (key == kthKey) keep = (atomicAdd(&tieCnt, 1) < kneed);
            const float v = keep ? fmaxf(mono2f(key), 0.0f) : 0.0f;
            ov[c4] = v;
            if (keep) {
                const int slot = atomicAdd(&slotCnt, 1);
                if (slot < TOPK) {
                    nzIdx[row * TOPK + slot] = e;
                    nzVal[row * TOPK + slot] = v;
                }
            }
        }
        f32x4 o; o.x = ov[0]; o.y = ov[1]; o.z = ov[2]; o.w = ov[3];
        __builtin_nontemporal_store(o, reinterpret_cast<f32x4*>(zrow) + tid + 1024 * j);
    }
}

// ============================================================
// Kernel C: sparse decode  h_hat[b,d] = b_dec[d] + sum_k val_k * W_dec[d, idx_k]
// W_dec gathers rely on L2 reuse across rows (columns shared batch-wide).
// ============================================================
__global__ __launch_bounds__(256) void sae_decode(
    const int* __restrict__ nzIdx, const float* __restrict__ nzVal,
    const float* __restrict__ W_dec, const float* __restrict__ b_dec,
    float* __restrict__ h_hat)
{
    const int row = blockIdx.x;
    const int tid = threadIdx.x;
    __shared__ int   sIdx[TOPK];
    __shared__ float sVal[TOPK];
    if (tid < TOPK) {
        sIdx[tid] = nzIdx[row * TOPK + tid];
        sVal[tid] = nzVal[row * TOPK + tid];
    }
    __syncthreads();

    float acc[16];
    #pragma unroll
    for (int j = 0; j < 16; ++j) acc[j] = b_dec[tid + 256 * j];

    for (int k = 0; k < TOPK; ++k) {
        const float v = sVal[k];
        const float* col = W_dec + sIdx[k];
        #pragma unroll
        for (int j = 0; j < 16; ++j)
            acc[j] = fmaf(v, col[(size_t)(tid + 256 * j) * LAT_DIM], acc[j]);
    }
    #pragma unroll
    for (int j = 0; j < 16; ++j)
        h_hat[(size_t)row * IN_DIM + tid + 256 * j] = acc[j];
}

// ============================================================
extern "C" void kernel_launch(void* const* d_in, const int* in_sizes, int n_in,
                              void* d_out, int out_size, void* d_ws, size_t ws_size,
                              hipStream_t stream) {
    (void)in_sizes; (void)n_in; (void)out_size; (void)ws_size;
    const float* h     = (const float*)d_in[0];
    const float* W_enc = (const float*)d_in[1];
    const float* b_enc = (const float*)d_in[2];
    const float* W_dec = (const float*)d_in[3];
    const float* b_dec = (const float*)d_in[4];

    float* out   = (float*)d_out;
    float* h_hat = out;                                   // [BATCH, IN_DIM]
    float* z     = out + (size_t)BATCH * IN_DIM;          // [BATCH, LAT_DIM]

    int*   nzIdx = (int*)d_ws;                                            // 1 MB
    float* nzVal = (float*)((char*)d_ws + (size_t)BATCH * TOPK * sizeof(int)); // 1 MB

    // x = M (fast): all-M blocks share each 2MB W_enc strip; W_enc read once.
    dim3 gridA(BATCH / BM, LAT_DIM / BN);                 // 32 x 256 workgroups
    sae_encode_gemm<<<gridA, 256, 0, stream>>>(h, W_enc, b_enc, z);
    sae_topk      <<<BATCH, 1024, 0, stream>>>(z, nzIdx, nzVal);
    sae_decode    <<<BATCH,  256, 0, stream>>>(nzIdx, nzVal, W_dec, b_dec, h_hat);
}
